// TopoConvLayer_72894184947742
// MI455X (gfx1250) — matrix operand order repaired and checked by reading, not compiled, not run on hardware
//
#include <hip/hip_runtime.h>
#include <hip/hip_bf16.h>
#include <math.h>

// ---------------------------------------------------------------------------
// TopoConvLayer fused CDNA5 (gfx1250) implementation.
// All GEMMs run on v_wmma_f32_16x16x32_f16 (f16 inputs, fp32 accumulate).
// This revision precomputes one lane-base pointer per memory stream and
// indexes with compile-time-constant offsets so stores/loads/atomics use
// instruction immediate offsets instead of per-element 64-bit VALU chains.
// ---------------------------------------------------------------------------

#define DEV __device__ __forceinline__

typedef __attribute__((ext_vector_type(16))) _Float16 v16h;
typedef __attribute__((ext_vector_type(8)))  _Float16 v8h;
typedef __attribute__((ext_vector_type(8)))  float    v8f;
typedef __attribute__((ext_vector_type(4)))  float    v4f;

constexpr int N_NODES   = 20000;
constexpr int N_EDGES   = 640000;
constexpr int DIM_NODE  = 256;   // also output dim of comb GEMM
constexpr int DIM_EDGE  = 128;
constexpr int N_CENTERS = 64;
constexpr int H         = 128;   // DIM_HIDDEN

// ---- fragment helpers (layouts per CDNA5 ISA §7.12.2) ----------

DEV v8h cvt8(const float* __restrict__ p) {
  v4f x = *(const v4f*)(p);
  v4f y = *(const v4f*)(p + 4);
  v8h r;
  r[0] = (_Float16)x[0]; r[1] = (_Float16)x[1];
  r[2] = (_Float16)x[2]; r[3] = (_Float16)x[3];
  r[4] = (_Float16)y[0]; r[5] = (_Float16)y[1];
  r[6] = (_Float16)y[2]; r[7] = (_Float16)y[3];
  return r;
}

DEV v16h cat16(v8h lo, v8h hi) {
  return __builtin_shufflevector(lo, hi, 0,1,2,3,4,5,6,7,8,9,10,11,12,13,14,15);
}

// A fragment (16x32 f16, M x K): lane holds row m=lane%16,
//   a[0..7]  = K0 + (lane/16)*8 + i ;  a[8..15] = K0 + 16 + (lane/16)*8 + i
// `row` must already include the +lhalf*8 lane offset; K0 is a constant.
DEV v16h a_frag_f32(const float* __restrict__ row, int K0) {
  return cat16(cvt8(row + K0), cvt8(row + K0 + 16));
}

DEV v16h a_frag_lds(const _Float16* row, int K0) {
  v8h lo = *(const v8h*)(row + K0);
  v8h hi = *(const v8h*)(row + K0 + 16);
  return cat16(lo, hi);
}

// B fragment (32x16 f16, K x N): lane holds column n = lane%16,
//   b[0..7] = K0 + (lane/16)*8 + i ;  b[8..15] = K0 + 16 + (lane/16)*8 + i
// `p` = Wt + n*Ktot + lhalf*8 + (constant nt*16*Ktot + K0)
DEV v16h b_frag_at(const _Float16* p) {
  v8h lo = *(const v8h*)(p);
  v8h hi = *(const v8h*)(p + 16);
  return cat16(lo, hi);
}

DEV v8f wmma_f16(v16h a, v16h b, v8f c) {
  return __builtin_amdgcn_wmma_f32_16x16x32_f16(
      /*neg_a=*/false, a, /*neg_b=*/false, b,
      /*c_mod=*/(short)0, c, /*reuse_a=*/false, /*reuse_b=*/false);
}

DEV v8f splat8(float v) {
  v8f c = {v, v, v, v, v, v, v, v};
  return c;
}

// ---------------------------------------------------------------------------
// Kernel 1: zero the scatter accumulator
// ---------------------------------------------------------------------------
__global__ void zero_kernel(float* __restrict__ p, int n) {
  for (int i = blockIdx.x * blockDim.x + threadIdx.x; i < n;
       i += gridDim.x * blockDim.x)
    p[i] = 0.0f;
}

// ---------------------------------------------------------------------------
// Kernel 2: node_proj = node_h @ W_node + b_node   [N_NODES x 128], fp32 out
// ---------------------------------------------------------------------------
__global__ __launch_bounds__(256) void node_proj_kernel(
    const float* __restrict__ node_h, const float* __restrict__ W_node,
    const float* __restrict__ b_node, float* __restrict__ np) {
  extern __shared__ char smem[];
  _Float16* Wt   = (_Float16*)smem;                 // [128][256] f16 (N-major)
  float*    bias = (float*)(smem + 128 * 256 * 2);  // [128]

  const int tid = threadIdx.x;
  for (int idx = tid; idx < DIM_NODE * H; idx += 256) {
    int k = idx / H, n = idx % H;                   // W_node[k][n], read coalesced
    Wt[n * DIM_NODE + k] = (_Float16)W_node[idx];
  }
  for (int idx = tid; idx < H; idx += 256) bias[idx] = b_node[idx];
  __syncthreads();

  const int wave = tid >> 5, lane = tid & 31;
  const int lrow = lane & 15, lhalf = lane >> 4;

  // lane-base pointers; all further indexing is by compile-time constants
  const _Float16* bp = Wt + (size_t)lrow * DIM_NODE + lhalf * 8;
  const float*    bl = bias + lrow;

  const int ntiles = N_NODES / 16;
  for (int t = blockIdx.x * 8 + wave; t < ntiles; t += gridDim.x * 8) {
    const int n0 = t * 16;
    const float* arow = node_h + (size_t)(n0 + lrow) * DIM_NODE + lhalf * 8;
    float*       ob   = np + (size_t)(n0 + lhalf * 8) * H + lrow;

    v16h a[8];
#pragma unroll
    for (int k = 0; k < 8; k++) a[k] = a_frag_f32(arow, k * 32);
#pragma unroll
    for (int nt = 0; nt < 8; nt++) {
      v8f c = splat8(bl[nt * 16]);
#pragma unroll
      for (int k = 0; k < 8; k++)
        c = wmma_f16(a[k], b_frag_at(bp + nt * 16 * DIM_NODE + k * 32), c);
#pragma unroll
      for (int i = 0; i < 8; i++)
        ob[i * H + nt * 16] = c[i];
    }
  }
}

// ---------------------------------------------------------------------------
// Kernel 3: fused per-edge pipeline + scatter-add
//   m1 = node_proj[src] * (dist @ W_dist + b_dist)
//   m2 = edge_h @ W_edge + b_edge
//   m  = tanh([m1|m2] @ W_comb); acc[dst] += m
// All weights live transposed in LDS as f16 (CDNA5: 320 KB LDS per WGP).
// ---------------------------------------------------------------------------
__global__ __launch_bounds__(256) void edge_kernel(
    const float* __restrict__ edge_h, const float* __restrict__ dist,
    const int* __restrict__ src, const int* __restrict__ dst,
    const float* __restrict__ W_edge, const float* __restrict__ b_edge,
    const float* __restrict__ W_dist, const float* __restrict__ b_dist,
    const float* __restrict__ W_comb, const float* __restrict__ np,
    float* __restrict__ acc) {
  extern __shared__ char smem[];
  _Float16* WtC   = (_Float16*)smem;          // [256][256]  128 KB
  _Float16* WtE   = WtC + 256 * 256;          // [128][128]   32 KB
  _Float16* WtD   = WtE + 128 * 128;          // [128][64]    16 KB
  _Float16* stage = WtD + 128 * 64;           // 8 waves x [16][256] 64 KB
  float* bE = (float*)(stage + 8 * 16 * 256); // [128]
  float* bD = bE + 128;                       // [128]

  const int tid = threadIdx.x;
  for (int idx = tid; idx < 256 * 256; idx += 256) {
    int k = idx / 256, n = idx % 256;
    WtC[n * 256 + k] = (_Float16)W_comb[idx];
  }
  for (int idx = tid; idx < 128 * 128; idx += 256) {
    int k = idx / 128, n = idx % 128;
    WtE[n * 128 + k] = (_Float16)W_edge[idx];
  }
  for (int idx = tid; idx < 64 * 128; idx += 256) {
    int k = idx / 128, n = idx % 128;
    WtD[n * 64 + k] = (_Float16)W_dist[idx];
  }
  for (int idx = tid; idx < 128; idx += 256) {
    bE[idx] = b_edge[idx];
    bD[idx] = b_dist[idx];
  }
  __syncthreads();

  const int wave = tid >> 5, lane = tid & 31;
  const int lrow = lane & 15, lhalf = lane >> 4;

  // lane-base pointers (loop-invariant)
  const _Float16* bpC = WtC + (size_t)lrow * 256 + lhalf * 8;
  const _Float16* bpE = WtE + (size_t)lrow * 128 + lhalf * 8;
  const _Float16* bpD = WtD + (size_t)lrow * 64  + lhalf * 8;
  const float*    bEl = bE + lrow;
  const float*    bDl = bD + lrow;
  _Float16* ms  = stage + wave * 16 * 256;               // wave's [16][256]
  _Float16* msw = ms + (lhalf * 8) * 256 + lrow;         // write base
  const _Float16* srow = ms + (size_t)lrow * 256 + lhalf * 8;  // A-read base

  const int ntiles = N_EDGES / 16;
  for (int t = blockIdx.x * 8 + wave; t < ntiles; t += gridDim.x * 8) {
    const int e0 = t * 16;

    // gather / scatter lane-base pointers for this lane's 8 C-fragment rows
    const float* gl[8];
    float*       ab[8];
#pragma unroll
    for (int i = 0; i < 8; i++) {
      const int m = lhalf * 8 + i;
      gl[i] = np  + (size_t)src[e0 + m] * H        + lrow;
      ab[i] = acc + (size_t)dst[e0 + m] * DIM_NODE + lrow;
    }

    // ---- m1 = node_proj[src] * (dist @ W_dist + b_dist) -> stage[:,0:128] --
    const float* drow = dist + (size_t)(e0 + lrow) * N_CENTERS + lhalf * 8;
    v16h ad0 = a_frag_f32(drow, 0);
    v16h ad1 = a_frag_f32(drow, 32);
#pragma unroll
    for (int nt = 0; nt < 8; nt++) {
      v8f c = splat8(bDl[nt * 16]);
      c = wmma_f16(ad0, b_frag_at(bpD + nt * 16 * 64 + 0), c);
      c = wmma_f16(ad1, b_frag_at(bpD + nt * 16 * 64 + 32), c);
#pragma unroll
      for (int i = 0; i < 8; i++)
        msw[i * 256 + nt * 16] = (_Float16)(c[i] * gl[i][nt * 16]);
    }

    // ---- m2 = edge_h @ W_edge + b_edge -> stage[:,128:256] ---------------
    const float* erow = edge_h + (size_t)(e0 + lrow) * DIM_EDGE + lhalf * 8;
    v16h ae[4];
#pragma unroll
    for (int k = 0; k < 4; k++) ae[k] = a_frag_f32(erow, k * 32);
#pragma unroll
    for (int nt = 0; nt < 8; nt++) {
      v8f c = splat8(bEl[nt * 16]);
#pragma unroll
      for (int k = 0; k < 4; k++)
        c = wmma_f16(ae[k], b_frag_at(bpE + nt * 16 * 128 + k * 32), c);
#pragma unroll
      for (int i = 0; i < 8; i++)
        msw[i * 256 + 128 + nt * 16] = (_Float16)c[i];
    }

    // ---- m = tanh(stage @ W_comb); scatter-add into acc[dst] -------------
    // DS ops are in-order per wave, so the stage writes above are visible.
    v16h am[8];
#pragma unroll
    for (int k = 0; k < 8; k++) am[k] = a_frag_lds(srow, k * 32);
#pragma unroll
    for (int nt = 0; nt < 16; nt++) {
      v8f c = splat8(0.0f);
#pragma unroll
      for (int k = 0; k < 8; k++)
        c = wmma_f16(am[k], b_frag_at(bpC + nt * 16 * 256 + k * 32), c);
#pragma unroll
      for (int i = 0; i < 8; i++)
        unsafeAtomicAdd(ab[i] + nt * 16, tanhf(c[i]));  // global_atomic_add_f32
    }
  }
}

// ---------------------------------------------------------------------------
// Kernel 4: per-node L2 normalization (one block per node row)
// ---------------------------------------------------------------------------
__global__ __launch_bounds__(256) void l2norm_kernel(const float* __restrict__ acc,
                                                     float* __restrict__ out) {
  __shared__ float red[256];
  const int row = blockIdx.x, t = threadIdx.x;
  const float v = acc[(size_t)row * DIM_NODE + t];
  red[t] = v * v;
  __syncthreads();
#pragma unroll
  for (int s = 128; s > 0; s >>= 1) {
    if (t < s) red[t] += red[t + s];
    __syncthreads();
  }
  out[(size_t)row * DIM_NODE + t] = v / sqrtf(red[0]);
}

// ---------------------------------------------------------------------------
extern "C" void kernel_launch(void* const* d_in, const int* in_sizes, int n_in,
                              void* d_out, int out_size, void* d_ws, size_t ws_size,
                              hipStream_t stream) {
  const float* node_h = (const float*)d_in[0];
  const float* edge_h = (const float*)d_in[1];
  const float* dist   = (const float*)d_in[2];
  const int*   src    = (const int*)d_in[3];
  const int*   dst    = (const int*)d_in[4];
  const float* W_node = (const float*)d_in[5];
  const float* b_node = (const float*)d_in[6];
  const float* W_edge = (const float*)d_in[7];
  const float* b_edge = (const float*)d_in[8];
  const float* W_dist = (const float*)d_in[9];
  const float* b_dist = (const float*)d_in[10];
  const float* W_comb = (const float*)d_in[11];
  float* out = (float*)d_out;

  // workspace: [acc: N_NODES*256 f32][np: N_NODES*128 f32]  (~30.7 MB)
  float* acc = (float*)d_ws;
  float* np  = acc + (size_t)N_NODES * DIM_NODE;

  zero_kernel<<<2048, 256, 0, stream>>>(acc, N_NODES * DIM_NODE);

  const size_t lds_node = 128 * 256 * 2 + 128 * 4;  // ~64.5 KB
  node_proj_kernel<<<160, 256, lds_node, stream>>>(node_h, W_node, b_node, np);

  const size_t lds_edge =
      (size_t)(256 * 256 + 128 * 128 + 128 * 64 + 8 * 16 * 256) * 2 + 256 * 4;  // ~241 KB
  edge_kernel<<<512, 256, lds_edge, stream>>>(edge_h, dist, src, dst, W_edge,
                                              b_edge, W_dist, b_dist, W_comb,
                                              np, acc);

  l2norm_kernel<<<N_NODES, 256, 0, stream>>>(acc, out);
}
